// GAWADecoder_45423574123229
// MI455X (gfx1250) — matrix operand order repaired and checked by reading, not compiled, not run on hardware
//
#include <hip/hip_runtime.h>
#include <hip/hip_bf16.h>

// ---------------------------------------------------------------------------
// GAWA decoder for MI455X (gfx1250, wave32, WMMA bf16 16x16x32)
//
//  * softmax over length-1 KV => attention output is a per-row constant
//    attn_add[b,:], precomputed once.
//  * layer-0 input gates = time-invariant (eword part) + 256-entry vocab LUT.
//  * persistent decode kernel: 256 WGs x 16 batch rows, h-state LDS-resident
//    across all T=32 steps, all GEMMs via v_wmma_f32_16x16x32_bf16.
//  * A-fragments hoisted to VGPRs once per GEMM phase; weights stream from L2.
//  * activations via v_exp_f32 + raw v_rcp_f32 (no IEEE div expansion).
// ---------------------------------------------------------------------------

typedef __bf16 bf16_t;
typedef __attribute__((ext_vector_type(16))) __bf16 v16bf;
typedef __attribute__((ext_vector_type(8)))  float  v8f;

typedef union { v16bf v; bf16_t e[16]; } ABu;

#define B_TOT 4096
#define T_TOT 32
#define H_DIM 256
#define G3    768   // 3*H
#define E_DIM 768
#define CE_DIM 64
#define V_DIM 256

__device__ __forceinline__ float fast_rcp(float x) { return __builtin_amdgcn_rcpf(x); }
__device__ __forceinline__ float sigm(float x)      { return fast_rcp(1.f + __expf(-x)); }
__device__ __forceinline__ float tanh_fast(float x) { return 2.f * fast_rcp(1.f + __expf(-2.f * x)) - 1.f; }

// A-fragment (16x32 bf16, MxK) lane layout: lanes 0-15 / 16-31 both hold M=0..15;
// element j of lane: K = hi*8+j (j<8) else 16+hi*8+(j-8), hi = lane>>4.
__device__ __forceinline__ int aK(int lane, int j) {
    int h = lane >> 4;
    return (j < 8) ? (h * 8 + j) : (16 + h * 8 + (j - 8));
}
// inverse: element (m, kk[0..31]) -> position within the 512-element panel
__device__ __forceinline__ int aPos(int m, int kk) {
    int h, j;
    if (kk < 16) { h = kk >> 3; j = kk & 7; }
    else         { int k2 = kk - 16; h = k2 >> 3; j = 8 + (k2 & 7); }
    return (h * 16 + m) * 16 + j;
}

// ---------------------------------------------------------------------------
// Pack a weight matrix W (N_out rows x K_in cols, f32, row stride `rowStride`,
// starting at row `rowOff`, col `colOff`) into WMMA B-fragment panels:
// dst[((nt*KT+kt)*32 + lane)*16 + j] = bf16(W[n, k]),
//   n = nt*16 + lane%16, k = kt*32 + aK(lane, j)   (B is K x N = W^T)
// Each lane's 16 bf16 = one contiguous 32B chunk -> global_load_b128 pairs.
// ---------------------------------------------------------------------------
__global__ __launch_bounds__(256) void pack_w_kernel(
    const float* __restrict__ src, bf16_t* __restrict__ dst,
    int K, int N, int rowStride, int rowOff, int colOff)
{
    int tid = blockIdx.x * 256 + threadIdx.x;
    if (tid >= K * N) return;
    int j     = tid & 15;
    int lane  = (tid >> 4) & 31;
    int panel = tid >> 9;
    int KT    = K >> 5;
    int kt    = panel % KT;
    int nt    = panel / KT;
    int n     = nt * 16 + (lane & 15);
    int k     = kt * 32 + aK(lane, j);
    dst[tid] = (bf16_t)src[(size_t)(rowOff + n) * rowStride + colOff + k];
}

// embgi[v, n] = sum_c char_emb[v,c] * gru_w_ih0[n, c]   (c < 64, n < 768)
__global__ __launch_bounds__(256) void embgi_kernel(
    const float* __restrict__ char_emb, const float* __restrict__ wih0,
    float* __restrict__ dst)
{
    int tid = blockIdx.x * 256 + threadIdx.x;   // 256*768 threads
    int v = tid / G3, n = tid - v * G3;
    const float* ce = char_emb + v * CE_DIM;
    const float* w  = wih0 + (size_t)n * (CE_DIM + E_DIM);
    float s = 0.f;
    #pragma unroll
    for (int c = 0; c < CE_DIM; ++c) s += ce[c] * w[c];
    dst[tid] = s;
}

// ---------------------------------------------------------------------------
// Generic WMMA GEMM for precompute: C = act(A @ B + bias), A f32 row-major
// (M x K, leading dim lda), B pre-packed bf16 panels (K x N), C f32 (M x N).
// One wave per 16x16 output tile. act: 0 = none, 1 = tanh.
// ---------------------------------------------------------------------------
__global__ __launch_bounds__(32) void gemm_bf16_kernel(
    const float* __restrict__ A, int lda,
    const bf16_t* __restrict__ Bp,
    const float* __restrict__ bias,
    float* __restrict__ C,
    int M, int N, int K, int act)
{
    int mt   = blockIdx.x % (M >> 4);
    int nt   = blockIdx.x / (M >> 4);
    int lane = threadIdx.x;
    int col  = lane & 15, hi = lane >> 4;
    int KT   = K >> 5;
    v8f acc = {};
    const float* ar = A + (size_t)(mt * 16 + col) * lda;
    for (int kt = 0; kt < KT; ++kt) {
        ABu au;
        const float* ap = ar + kt * 32;
        #pragma unroll
        for (int j = 0; j < 16; ++j) au.e[j] = (bf16_t)ap[aK(lane, j)];
        v16bf b = *(const v16bf*)(Bp + ((size_t)(nt * KT + kt)) * 512 + lane * 16);
        acc = __builtin_amdgcn_wmma_f32_16x16x32_bf16(
            false, au.v, false, b, (short)0, acc, false, false);
    }
    float bb = bias[nt * 16 + col];
    #pragma unroll
    for (int r = 0; r < 8; ++r) {
        float v = acc[r] + bb;
        if (act) v = tanh_fast(v);
        C[(size_t)(mt * 16 + r + hi * 8) * N + nt * 16 + col] = v;
    }
}

// ---------------------------------------------------------------------------
// Persistent decode kernel: 1 workgroup = 16 batch rows, 8 waves, loops T steps.
// LDS: h0/h1 state (f32 + packed bf16 A-frags), gate staging (bf16),
// packed "out" A-frag. 104 KB dynamic LDS -> 3 WGs per 320KB WGP.
// ---------------------------------------------------------------------------
__global__ __launch_bounds__(256) void gawad_decode_kernel(
    const int*   __restrict__ tids,      // (B, T) teacher-forced targets
    const float* __restrict__ h0init,    // (B, 256) tanh(eword@Wp^T+b)
    const float* __restrict__ gi0s,      // (B, 768) static input gates L0 (incl b_ih0)
    const float* __restrict__ embgi,     // (V, 768) vocab LUT for L0 input gates
    const float* __restrict__ attn_add,  // (B, 256) constant attention output
    const float* __restrict__ bhh0,      // (768,)
    const float* __restrict__ bih1,      // (768,)
    const float* __restrict__ bhh1,      // (768,)
    const float* __restrict__ projb,     // (256,)
    const bf16_t* __restrict__ whh0p,    // packed (256->768)
    const bf16_t* __restrict__ wih1p,    // packed (256->768)
    const bf16_t* __restrict__ whh1p,    // packed (256->768)
    const bf16_t* __restrict__ projp,    // packed (256->256)
    float* __restrict__ out)             // (B, T, 256)
{
    extern __shared__ unsigned char smem[];
    bf16_t* h0p  = (bf16_t*)smem;                   // 16x256 A-frag panels (8x512)
    bf16_t* h1p  = h0p  + 4096;
    bf16_t* outp = h1p  + 4096;
    bf16_t* gib  = outp + 4096;                     // 16x768 bf16 gate staging
    bf16_t* ghb  = gib  + 12288;
    float* h0f = (float*)(smem + (size_t)(3 * 4096 + 2 * 12288) * sizeof(bf16_t));
    float* h1f = h0f + 4096;                        // 16x256 f32 each

    const int tid  = threadIdx.x;
    const int lane = tid & 31;
    const int wave = tid >> 5;
    const int col  = lane & 15;
    const int hi   = lane >> 4;
    const int b0   = blockIdx.x * 16;

    // ---- prologue: h0 = h1 = h0init tile -------------------------------------
    for (int i = 0; i < 16; ++i) {
        int e = tid + (i << 8);
        int m = e >> 8, n = e & 255;
        float v = h0init[(size_t)(b0 + m) * H_DIM + n];
        h0f[e] = v; h1f[e] = v;
        bf16_t bv = (bf16_t)v;
        int p = (n >> 5) * 512 + aPos(m, n & 31);
        h0p[p] = bv; h1p[p] = bv;
    }
    __syncthreads();

    for (int t = 0; t < T_TOT; ++t) {
        // ---- GEMM1: ghb = h0_old @ Whh0^T  (48 tiles over 8 waves) ----------
        {
            v16bf af[8];
            #pragma unroll
            for (int kt = 0; kt < 8; ++kt)
                af[kt] = *(const v16bf*)(h0p + kt * 512 + lane * 16);
            for (int nt = wave; nt < 48; nt += 8) {
                if (nt + 8 < 48)
                    __builtin_prefetch(whh0p + ((size_t)(nt + 8) * 8) * 512, 0, 1);
                v8f acc = {};
                #pragma unroll
                for (int kt = 0; kt < 8; ++kt) {
                    v16bf b = *(const v16bf*)(whh0p + ((size_t)(nt * 8 + kt)) * 512 + lane * 16);
                    acc = __builtin_amdgcn_wmma_f32_16x16x32_bf16(
                        false, af[kt], false, b, (short)0, acc, false, false);
                }
                #pragma unroll
                for (int r = 0; r < 8; ++r)
                    ghb[(r + hi * 8) * G3 + nt * 16 + col] = (bf16_t)acc[r];
            }
        }
        __syncthreads();

        // ---- GRU layer 0 elementwise ---------------------------------------
        for (int i = 0; i < 16; ++i) {
            int e = tid + (i << 8);
            int m = e >> 8, n = e & 255;
            int brow = b0 + m;
            int id = (t == 0) ? 1 : tids[(size_t)brow * T_TOT + t - 1];  // BOS=1
            const float* gs = gi0s + (size_t)brow * G3;
            const float* ge = embgi + (size_t)id * G3;
            float gir = gs[n]       + ge[n];
            float giz = gs[n + 256] + ge[n + 256];
            float gin = gs[n + 512] + ge[n + 512];
            float ghr = (float)ghb[m * G3 + n]       + bhh0[n];
            float ghz = (float)ghb[m * G3 + n + 256] + bhh0[n + 256];
            float ghn = (float)ghb[m * G3 + n + 512] + bhh0[n + 512];
            float r  = sigm(gir + ghr);
            float z  = sigm(giz + ghz);
            float nn = tanh_fast(gin + r * ghn);
            float h  = h0f[e];
            float hn = (1.f - z) * nn + z * h;
            h0f[e] = hn;
            h0p[(n >> 5) * 512 + aPos(m, n & 31)] = (bf16_t)hn;
        }
        __syncthreads();

        // ---- GEMM2a: gib = h0_new @ Wih1^T ----------------------------------
        {
            v16bf af[8];
            #pragma unroll
            for (int kt = 0; kt < 8; ++kt)
                af[kt] = *(const v16bf*)(h0p + kt * 512 + lane * 16);
            for (int nt = wave; nt < 48; nt += 8) {
                v8f acc = {};
                #pragma unroll
                for (int kt = 0; kt < 8; ++kt) {
                    v16bf b = *(const v16bf*)(wih1p + ((size_t)(nt * 8 + kt)) * 512 + lane * 16);
                    acc = __builtin_amdgcn_wmma_f32_16x16x32_bf16(
                        false, af[kt], false, b, (short)0, acc, false, false);
                }
                #pragma unroll
                for (int r = 0; r < 8; ++r)
                    gib[(r + hi * 8) * G3 + nt * 16 + col] = (bf16_t)acc[r];
            }
        }
        // ---- GEMM2b: ghb = h1_old @ Whh1^T ----------------------------------
        {
            v16bf af[8];
            #pragma unroll
            for (int kt = 0; kt < 8; ++kt)
                af[kt] = *(const v16bf*)(h1p + kt * 512 + lane * 16);
            for (int nt = wave; nt < 48; nt += 8) {
                v8f acc = {};
                #pragma unroll
                for (int kt = 0; kt < 8; ++kt) {
                    v16bf b = *(const v16bf*)(whh1p + ((size_t)(nt * 8 + kt)) * 512 + lane * 16);
                    acc = __builtin_amdgcn_wmma_f32_16x16x32_bf16(
                        false, af[kt], false, b, (short)0, acc, false, false);
                }
                #pragma unroll
                for (int r = 0; r < 8; ++r)
                    ghb[(r + hi * 8) * G3 + nt * 16 + col] = (bf16_t)acc[r];
            }
        }
        __syncthreads();

        // ---- GRU layer 1 elementwise + attn_add -> packed "out" A-frag -----
        for (int i = 0; i < 16; ++i) {
            int e = tid + (i << 8);
            int m = e >> 8, n = e & 255;
            int brow = b0 + m;
            float gir = (float)gib[m * G3 + n]       + bih1[n];
            float giz = (float)gib[m * G3 + n + 256] + bih1[n + 256];
            float gin = (float)gib[m * G3 + n + 512] + bih1[n + 512];
            float ghr = (float)ghb[m * G3 + n]       + bhh1[n];
            float ghz = (float)ghb[m * G3 + n + 256] + bhh1[n + 256];
            float ghn = (float)ghb[m * G3 + n + 512] + bhh1[n + 512];
            float r  = sigm(gir + ghr);
            float z  = sigm(giz + ghz);
            float nn = tanh_fast(gin + r * ghn);
            float h  = h1f[e];
            float hn = (1.f - z) * nn + z * h;
            h1f[e] = hn;
            int p = (n >> 5) * 512 + aPos(m, n & 31);
            h1p[p] = (bf16_t)hn;
            float ov = hn + attn_add[(size_t)brow * H_DIM + n];
            outp[p] = (bf16_t)ov;
        }
        __syncthreads();

        // ---- GEMM3: logits = out @ projW^T + proj_b -> global ---------------
        {
            v16bf af[8];
            #pragma unroll
            for (int kt = 0; kt < 8; ++kt)
                af[kt] = *(const v16bf*)(outp + kt * 512 + lane * 16);
            for (int nt = wave; nt < 16; nt += 8) {
                v8f acc = {};
                #pragma unroll
                for (int kt = 0; kt < 8; ++kt) {
                    v16bf b = *(const v16bf*)(projp + ((size_t)(nt * 8 + kt)) * 512 + lane * 16);
                    acc = __builtin_amdgcn_wmma_f32_16x16x32_bf16(
                        false, af[kt], false, b, (short)0, acc, false, false);
                }
                float bb = projb[nt * 16 + col];
                #pragma unroll
                for (int r = 0; r < 8; ++r) {
                    size_t o = ((size_t)(b0 + r + hi * 8) * T_TOT + t) * V_DIM + nt * 16 + col;
                    out[o] = acc[r] + bb;
                }
            }
        }
        __syncthreads();
    }
}

// ---------------------------------------------------------------------------
// Host launcher
// ---------------------------------------------------------------------------
extern "C" void kernel_launch(void* const* d_in, const int* in_sizes, int n_in,
                              void* d_out, int out_size, void* d_ws, size_t ws_size,
                              hipStream_t stream) {
    (void)in_sizes; (void)n_in; (void)out_size; (void)ws_size;
    const float* eword      = (const float*)d_in[0];
    const int*   tids       = (const int*)  d_in[1];
    const float* char_emb   = (const float*)d_in[2];
    const float* ewproj_w   = (const float*)d_in[3];
    const float* ewproj_b   = (const float*)d_in[4];
    const float* wih0       = (const float*)d_in[5];
    const float* whh0       = (const float*)d_in[6];
    const float* bih0       = (const float*)d_in[7];
    const float* bhh0       = (const float*)d_in[8];
    const float* wih1       = (const float*)d_in[9];
    const float* whh1       = (const float*)d_in[10];
    const float* bih1       = (const float*)d_in[11];
    const float* bhh1       = (const float*)d_in[12];
    const float* attn_in_w  = (const float*)d_in[13];
    const float* attn_in_b  = (const float*)d_in[14];
    const float* attn_out_w = (const float*)d_in[15];
    const float* attn_out_b = (const float*)d_in[16];
    // d_in[17..18] = key_w/key_b: unused (length-1 softmax => weights == 1)
    const float* val_w      = (const float*)d_in[19];
    const float* val_b      = (const float*)d_in[20];
    const float* proj_w     = (const float*)d_in[21];
    const float* proj_b     = (const float*)d_in[22];
    float* out = (float*)d_out;

    // workspace carving (256B aligned)
    unsigned char* ws = (unsigned char*)d_ws;
    size_t off = 0;
    auto take = [&](size_t bytes) -> unsigned char* {
        unsigned char* p = ws + off;
        off = (off + bytes + 255) & ~(size_t)255;
        return p;
    };
    bf16_t* whh0p   = (bf16_t*)take((size_t)256 * 768 * 2);
    bf16_t* wih1p   = (bf16_t*)take((size_t)256 * 768 * 2);
    bf16_t* whh1p   = (bf16_t*)take((size_t)256 * 768 * 2);
    bf16_t* projp   = (bf16_t*)take((size_t)256 * 256 * 2);
    bf16_t* ewprojp = (bf16_t*)take((size_t)768 * 256 * 2);
    bf16_t* wih0Ep  = (bf16_t*)take((size_t)768 * 768 * 2);
    bf16_t* valp    = (bf16_t*)take((size_t)768 * 256 * 2);
    bf16_t* wvp     = (bf16_t*)take((size_t)256 * 256 * 2);
    bf16_t* aowp    = (bf16_t*)take((size_t)256 * 256 * 2);
    float* embgi    = (float*)take((size_t)V_DIM * G3 * 4);
    float* h0init   = (float*)take((size_t)B_TOT * H_DIM * 4);
    float* gi0s     = (float*)take((size_t)B_TOT * G3 * 4);
    float* t1       = (float*)take((size_t)B_TOT * H_DIM * 4);
    float* ev       = (float*)take((size_t)B_TOT * H_DIM * 4);
    float* attn_add = (float*)take((size_t)B_TOT * H_DIM * 4);

    // ---- pack all GEMM weights into WMMA B-fragment panels (bf16) ----------
    auto pack = [&](const float* src, bf16_t* dst, int K, int N,
                    int rs, int ro, int co) {
        pack_w_kernel<<<(K * N) / 256, 256, 0, stream>>>(src, dst, K, N, rs, ro, co);
    };
    pack(whh0,       whh0p,   256, 768, 256, 0, 0);
    pack(wih1,       wih1p,   256, 768, 256, 0, 0);
    pack(whh1,       whh1p,   256, 768, 256, 0, 0);
    pack(proj_w,     projp,   256, 256, 256, 0, 0);
    pack(ewproj_w,   ewprojp, 768, 256, 768, 0, 0);
    pack(wih0,       wih0Ep,  768, 768, 832, 0, 64);   // eword-facing cols of W_ih0
    pack(val_w,      valp,    768, 256, 768, 0, 0);
    pack(attn_in_w,  wvp,     256, 256, 256, 512, 0);  // wv = rows 512..767
    pack(attn_out_w, aowp,    256, 256, 256, 0, 0);

    // ---- vocab LUT for layer-0 input gates ---------------------------------
    embgi_kernel<<<(V_DIM * G3) / 256, 256, 0, stream>>>(char_emb, wih0, embgi);

    // ---- precompute GEMMs (all via WMMA) -----------------------------------
    // h0init = tanh(eword @ ewproj^T + b)
    gemm_bf16_kernel<<<(B_TOT / 16) * (H_DIM / 16), 32, 0, stream>>>(
        eword, E_DIM, ewprojp, ewproj_b, h0init, B_TOT, H_DIM, E_DIM, 1);
    // gi0s = eword @ W_ih0[:,CE:]^T + b_ih0
    gemm_bf16_kernel<<<(B_TOT / 16) * (G3 / 16), 32, 0, stream>>>(
        eword, E_DIM, wih0Ep, bih0, gi0s, B_TOT, G3, E_DIM, 0);
    // t1 = eword @ val_w^T + val_b
    gemm_bf16_kernel<<<(B_TOT / 16) * (H_DIM / 16), 32, 0, stream>>>(
        eword, E_DIM, valp, val_b, t1, B_TOT, H_DIM, E_DIM, 0);
    // ev = t1 @ wv^T + bv
    gemm_bf16_kernel<<<(B_TOT / 16) * (H_DIM / 16), 32, 0, stream>>>(
        t1, H_DIM, wvp, attn_in_b + 512, ev, B_TOT, H_DIM, H_DIM, 0);
    // attn_add = ev @ attn_out_w^T + attn_out_b
    gemm_bf16_kernel<<<(B_TOT / 16) * (H_DIM / 16), 32, 0, stream>>>(
        ev, H_DIM, aowp, attn_out_b, attn_add, B_TOT, H_DIM, H_DIM, 0);

    // ---- persistent decode -------------------------------------------------
    size_t lds = (size_t)(3 * 4096 + 2 * 12288) * 2 + (size_t)2 * 4096 * 4; // 106496 B
    (void)hipFuncSetAttribute((const void*)gawad_decode_kernel,
                              hipFuncAttributeMaxDynamicSharedMemorySize, (int)lds);
    gawad_decode_kernel<<<B_TOT / 16, 256, lds, stream>>>(
        tids, h0init, gi0s, embgi, attn_add, bhh0, bih1, bhh1, proj_b,
        whh0p, wih1p, whh1p, projp, out);
}
